// MultiHeadAttention_82729660056256
// MI455X (gfx1250) — matrix operand order repaired and checked by reading, not compiled, not run on hardware
//
#include <hip/hip_runtime.h>

typedef __attribute__((ext_vector_type(16))) _Float16 v16h;
typedef __attribute__((ext_vector_type(8)))  _Float16 v8h;
typedef __attribute__((ext_vector_type(2)))  __fp16   v2h;   // matches __builtin_amdgcn_cvt_pkrtz return type
typedef __attribute__((ext_vector_type(8)))  float    v8f;

namespace {
constexpr int   kH    = 16;
constexpr int   kDK   = 64;
constexpr int   kDV   = 64;
constexpr int   kL    = 2048;
constexpr int   kBQ   = 16;      // query rows per wave
constexpr int   kWAVES= 4;       // waves per block
constexpr int   kBK   = 64;      // keys per iteration (two 32-key contraction slices)
constexpr float kScale    = 0.125f;               // 1/sqrt(64), exact power of two
constexpr float kMaskFill = -1.0e10f * 0.125f;    // matches reference: mask fill then /8
constexpr float kLog2e    = 1.44269504088896340736f;
}

static __device__ inline v16h combine16(v8h lo, v8h hi) {
  v16h r;
#pragma unroll
  for (int i = 0; i < 8; ++i) { r[i] = lo[i]; r[i + 8] = hi[i]; }
  return r;
}

// ---- VALU-only 16-lane max reduction via DPP16 ----
template <int CTRL>
static __device__ inline float dpp_mov(float x) {
  int r = __builtin_amdgcn_update_dpp(0, __float_as_int(x), CTRL, 0xF, 0xF, true);
  return __int_as_float(r);
}
static __device__ inline float red_max16(float x) {
  x = fmaxf(x, dpp_mov<0xB1>(x));   // quad_perm(1,0,3,2)
  x = fmaxf(x, dpp_mov<0x4E>(x));   // quad_perm(2,3,0,1)
  x = fmaxf(x, dpp_mov<0x141>(x));  // row_half_mirror
  x = fmaxf(x, dpp_mov<0x140>(x));  // row_mirror
  return x;
}

__global__ __launch_bounds__(kWAVES * 32)
void mha_flash_wmma_kernel(const float* __restrict__ Q, const float* __restrict__ K,
                           const float* __restrict__ V, const unsigned char* __restrict__ mask,
                           float* __restrict__ out) {
  // P and V use a per-32-slice permuted contraction order c = slice*32 + 2*(key&15) + ((key>>4)&1).
  // The same relabeling on both operands leaves P*V unchanged; it makes P stores packed b32 and
  // keeps every fragment gather a contiguous 16-byte LDS load.
  __shared__ __attribute__((aligned(16))) _Float16      sK [kBK][kDK];         // 8 KB (key-major, natural)
  __shared__ __attribute__((aligned(16))) _Float16      sVt[kDV][kBK];         // 8 KB (dv-major, c-order)
  __shared__ __attribute__((aligned(16))) _Float16      sQ [kWAVES][kBQ][kDK]; // 8 KB (pre-scaled by 1/8)
  __shared__ __attribute__((aligned(16))) _Float16      sP [kWAVES][kBQ][kBK]; // 8 KB (c-order)
  __shared__ __attribute__((aligned(16))) unsigned char sM [kWAVES * kBQ][kBK];// 4 KB mask tile

  const int tid  = threadIdx.x;
  const int lane = tid & 31;
  const int wv   = tid >> 5;
  const int half = lane >> 4;   // 0: lanes 0-15, 1: lanes 16-31
  const int l16  = lane & 15;

  const int h   = blockIdx.y;
  const int b   = blockIdx.z;
  const int qb0 = blockIdx.x * (kWAVES * kBQ);
  const int qb  = qb0 + wv * kBQ;

  const long qkStride = (long)kH * kDK;
  const long vStride  = (long)kH * kDV;

  // ---- stage this wave's 16x64 Q tile, fp32 -> fp16, pre-scaled by 1/8 (exact) ----
  {
    const float* qbase = Q + ((long)b * kL + qb) * qkStride + (long)h * kDK;
    for (int i = lane; i < kBQ * kDK / 2; i += 32) {
      int r = i >> 5, d = (i & 31) * 2;
      float2 qf = *(const float2*)(qbase + (long)r * qkStride + d);
      *(v2h*)&sQ[wv][r][d] = __builtin_amdgcn_cvt_pkrtz(qf.x * kScale, qf.y * kScale);
    }
  }
  __syncthreads();

  // Q A-fragments: two 16x32 slices (dims 0..31 and 32..63)
  v16h aq0, aq1;
  {
    const _Float16* qrow = &sQ[wv][l16][0];
    v8h x0 = *(const v8h*)(qrow +      half * 8);
    v8h x1 = *(const v8h*)(qrow + 16 + half * 8);
    v8h x2 = *(const v8h*)(qrow + 32 + half * 8);
    v8h x3 = *(const v8h*)(qrow + 48 + half * 8);
    aq0 = combine16(x0, x1);
    aq1 = combine16(x2, x3);
  }

  // all-ones B fragment: P x ones -> per-row sums broadcast across lanes
  v16h vones;
#pragma unroll
  for (int i = 0; i < 16; ++i) vones[i] = (_Float16)1.0f;

  // online-softmax state: 8 rows per lane (row = vx + 8*half)
  float mi[8];
  v8f li = {};
  v8f o[4] = {{}, {}, {}, {}};
#pragma unroll
  for (int i = 0; i < 8; ++i) mi[i] = -3.0e38f;

  const unsigned char* mblockbase = mask + ((long)b * kL + qb0) * kL;

  for (int kb = 0; kb < kL / kBK; ++kb) {
    const int k0 = kb * kBK;
    __syncthreads();   // previous iteration's LDS reads complete before overwrite

    // ---- cooperative stage of K (natural), V (transposed, c-order), mask; fp32 -> fp16 ----
    {
      const float* kbase = K + ((long)b * kL + k0) * qkStride + (long)h * kDK;
      const float* vbase = V + ((long)b * kL + k0) * vStride  + (long)h * kDV;
#pragma unroll
      for (int j = 0; j < 16; ++j) {
        int idx = tid + j * 128;                      // 0..2047
        // K: pair along d
        int kr = idx >> 5, kd = (idx & 31) * 2;
        float2 kf = *(const float2*)(kbase + (long)kr * qkStride + kd);
        *(v2h*)&sK[kr][kd] = __builtin_amdgcn_cvt_pkrtz(kf.x, kf.y);
        // V: pair along key within a 32-slice (keys r and r+16 adjacent in c-order)
        int vd = idx & 63, vrp = idx >> 6;            // 0..31
        int slice = vrp >> 4, rl = vrp & 15;
        int vkey  = slice * 32 + rl;
        float va = vbase[(long)vkey        * vStride + vd];
        float vb = vbase[(long)(vkey + 16) * vStride + vd];
        *(v2h*)&sVt[vd][slice * 32 + 2 * rl] = __builtin_amdgcn_cvt_pkrtz(va, vb);
      }
      // coalesced mask tile: 64 rows x 64 bytes; two uint4 per thread
      const int mr = tid >> 1, mo = (tid & 1) * 32;
      *(uint4*)&sM[mr][mo]      = *(const uint4*)(mblockbase + (long)mr * kL + k0 + mo);
      *(uint4*)&sM[mr][mo + 16] = *(const uint4*)(mblockbase + (long)mr * kL + k0 + mo + 16);
    }
    __syncthreads();

    // ---- S(16x64) = Qscaled(16x64) * K_block^T : 4 n-tiles x 2 k-steps ----
    v8f s[4];
#pragma unroll
    for (int t = 0; t < 4; ++t) {
      const _Float16* kr = &sK[t * 16 + l16][0];    // B col n -> key = t*16 + n
      v16h bA = combine16(*(const v8h*)(kr +      half * 8), *(const v8h*)(kr + 16 + half * 8));
      v16h bB = combine16(*(const v8h*)(kr + 32 + half * 8), *(const v8h*)(kr + 48 + half * 8));
      v8f acc = {};
      acc  = __builtin_amdgcn_wmma_f32_16x16x32_f16(false, aq0, false, bA, (short)0, acc, false, false);
      s[t] = __builtin_amdgcn_wmma_f32_16x16x32_f16(false, aq1, false, bB, (short)0, acc, false, false);
    }

    // ---- mask (pre-scale semantics preserved) + online softmax over 64 keys ----
    v8f corrv;
#pragma unroll
    for (int vx = 0; vx < 8; ++vx) {
      const int m_ = vx + half * 8;
      const unsigned char* mp = &sM[wv * kBQ + m_][l16];
      float a0 = mp[0]  ? kMaskFill : s[0][vx];     // key = k0 + l16
      float a1 = mp[16] ? kMaskFill : s[1][vx];     // key = k0 + 16 + l16
      float a2 = mp[32] ? kMaskFill : s[2][vx];     // key = k0 + 32 + l16
      float a3 = mp[48] ? kMaskFill : s[3][vx];     // key = k0 + 48 + l16

      float mn = fmaxf(mi[vx], red_max16(fmaxf(fmaxf(a0, a1), fmaxf(a2, a3))));
      corrv[vx] = __builtin_exp2f((mi[vx] - mn) * kLog2e);
      mi[vx] = mn;
      float p0 = __builtin_exp2f((a0 - mn) * kLog2e);
      float p1 = __builtin_exp2f((a1 - mn) * kLog2e);
      float p2 = __builtin_exp2f((a2 - mn) * kLog2e);
      float p3 = __builtin_exp2f((a3 - mn) * kLog2e);
      // packed c-order stores: slice0 cols (2*l16, 2*l16+1), slice1 cols (32+2*l16, 33+2*l16)
      *(v2h*)&sP[wv][m_][2 * l16]      = __builtin_amdgcn_cvt_pkrtz(p0, p1);
      *(v2h*)&sP[wv][m_][32 + 2 * l16] = __builtin_amdgcn_cvt_pkrtz(p2, p3);
    }
    o[0] *= corrv; o[1] *= corrv; o[2] *= corrv; o[3] *= corrv;
    __syncthreads();

    // ---- O(16x64) += P(16x64) * V_block(64x64); row sums via P x ones ----
    {
      const _Float16* pr = &sP[wv][l16][0];
      v16h ap0 = combine16(*(const v8h*)(pr +      half * 8), *(const v8h*)(pr + 16 + half * 8));
      v16h ap1 = combine16(*(const v8h*)(pr + 32 + half * 8), *(const v8h*)(pr + 48 + half * 8));
      v8f rsum = {};
      rsum = __builtin_amdgcn_wmma_f32_16x16x32_f16(false, ap0, false, vones, (short)0, rsum, false, false);
      rsum = __builtin_amdgcn_wmma_f32_16x16x32_f16(false, ap1, false, vones, (short)0, rsum, false, false);
#pragma unroll
      for (int g = 0; g < 4; ++g) {
        const _Float16* vt = &sVt[g * 16 + l16][0];   // B col n -> dv = g*16 + n
        v16h b0 = combine16(*(const v8h*)(vt +      half * 8), *(const v8h*)(vt + 16 + half * 8));
        v16h b1 = combine16(*(const v8h*)(vt + 32 + half * 8), *(const v8h*)(vt + 48 + half * 8));
        o[g] = __builtin_amdgcn_wmma_f32_16x16x32_f16(false, ap0, false, b0, (short)0, o[g], false, false);
        o[g] = __builtin_amdgcn_wmma_f32_16x16x32_f16(false, ap1, false, b1, (short)0, o[g], false, false);
      }
      li = li * corrv + rsum;
    }
  }

  // ---- epilogue: normalize and store ----
  float* obase = out + ((long)b * kL + qb) * vStride + (long)h * kDV;
#pragma unroll
  for (int vx = 0; vx < 8; ++vx) {
    const int m_ = vx + half * 8;
    float inv = 1.0f / li[vx];
    float* orow = obase + (long)m_ * vStride + l16;
    orow[0]  = o[0][vx] * inv;
    orow[16] = o[1][vx] * inv;
    orow[32] = o[2][vx] * inv;
    orow[48] = o[3][vx] * inv;
  }
}

extern "C" void kernel_launch(void* const* d_in, const int* in_sizes, int n_in,
                              void* d_out, int out_size, void* d_ws, size_t ws_size,
                              hipStream_t stream) {
  const float*         Q    = (const float*)d_in[0];
  const float*         K    = (const float*)d_in[1];
  const float*         V    = (const float*)d_in[2];
  const unsigned char* mask = (const unsigned char*)d_in[3];   // jnp.bool_: 1 byte/elem
  float*               out  = (float*)d_out;

  const int B = in_sizes[0] / (kL * kH * kDK);   // = 2
  dim3 grid(kL / (kBQ * kWAVES), kH, B);         // (32, 16, 2)
  dim3 block(kWAVES * 32);                       // 4 waves of 32
  hipLaunchKernelGGL(mha_flash_wmma_kernel, grid, block, 0, stream, Q, K, V, mask, out);
}
